// CausalTimeSeriesMamba_60284160967203
// MI455X (gfx1250) — compile-verified, matrix operand order checked by reading, not run on hardware
//
#include <hip/hip_runtime.h>
#include <hip/hip_bf16.h>

typedef __bf16 bf16;
typedef __bf16 bf16x8  __attribute__((ext_vector_type(8)));
typedef __bf16 bf16x16 __attribute__((ext_vector_type(16)));
typedef float  f32x8   __attribute__((ext_vector_type(8)));

#define DEVINL __device__ __forceinline__

// ---------------- model constants ----------------
constexpr int Bc  = 2;
constexpr int Tc  = 1024;
constexpr int Fc  = 64;
constexpr int Dc  = 512;
constexpr int Lc  = 4;
constexpr int SN  = 16;          // state dim N
constexpr int CK  = 4;           // conv taps K
constexpr int FFc = 2048;
constexpr int DIc = 2 * Dc;      // 1024
constexpr int Rc  = Dc / 16;     // 32
constexpr int Mc  = Bc * Tc;     // 2048 rows
constexpr int NRc = Rc + 2 * SN; // 64

DEVINL bf16 f2bf(float f) {
  union { float f; unsigned u; } v; v.f = f;
  unsigned r = v.u + 0x7FFFu + ((v.u >> 16) & 1u);   // round-to-nearest-even
  unsigned short h = (unsigned short)(r >> 16);
  return __builtin_bit_cast(bf16, h);
}

DEVINL bf16x16 load_frag(const bf16* row, int k, int half) {
  bf16x8 lo = *(const bf16x8*)(row + k + 8 * half);
  bf16x8 hi = *(const bf16x8*)(row + k + 16 + 8 * half);
  bf16x16 v;
#pragma unroll
  for (int i = 0; i < 8; ++i) { v[i] = lo[i]; v[i + 8] = hi[i]; }
  return v;
}

// ---------------- f32 -> bf16 conversion ----------------
__global__ __launch_bounds__(256)
void cvt_bf16(const float* __restrict__ in, bf16* __restrict__ out, int n) {
  int i = blockIdx.x * 256 + threadIdx.x;
  if (i < n) out[i] = f2bf(in[i]);
}

// ---------------- WMMA GEMM: C[M,N] = epi(A[M,K] @ W[N,K]^T + bias) ----------------
// One wave computes a 16(M) x 64(N) strip: one A fragment feeds 4 WMMAs.
enum { EPI_NONE = 0, EPI_BIAS = 1, EPI_GELU = 2, EPI_SOFTPLUS = 3, EPI_RESID = 4 };

template<int EPI>
__global__ __launch_bounds__(256)
void wmma_gemm(const bf16* __restrict__ A, int lda,
               const bf16* __restrict__ W, int ldw,
               const float* __restrict__ bias,
               float* __restrict__ Cf, bf16* __restrict__ Cb, int ldc,
               int Mm, int Nn, int Kk)
{
  const int nstrips = Nn >> 6;                       // N / 64
  const int total   = (Mm >> 4) * nstrips;
  const int strip   = blockIdx.x * 8 + (threadIdx.x >> 5);
  if (strip >= total) return;                        // wave-uniform exit
  const int tm   = strip / nstrips;
  const int n0   = (strip - tm * nstrips) * 64;      // first output column
  const int lane = threadIdx.x & 31;
  const int half = lane >> 4;
  const int r    = lane & 15;

  const bf16* arow = A + (size_t)(tm * 16 + r) * lda;
  const bf16* wrow = W + (size_t)(n0 + r) * ldw;     // +u*16*ldw per N-tile

  f32x8 acc0 = {}, acc1 = {}, acc2 = {}, acc3 = {};
  for (int k = 0; k < Kk; k += 32) {
    bf16x16 av = load_frag(arow, k, half);
    bf16x16 b0 = load_frag(wrow + (size_t)0  * ldw, k, half);
    bf16x16 b1 = load_frag(wrow + (size_t)16 * ldw, k, half);
    bf16x16 b2 = load_frag(wrow + (size_t)32 * ldw, k, half);
    bf16x16 b3 = load_frag(wrow + (size_t)48 * ldw, k, half);
    acc0 = __builtin_amdgcn_wmma_f32_16x16x32_bf16(false, av, false, b0, (short)0, acc0, false, false);
    acc1 = __builtin_amdgcn_wmma_f32_16x16x32_bf16(false, av, false, b1, (short)0, acc1, false, false);
    acc2 = __builtin_amdgcn_wmma_f32_16x16x32_bf16(false, av, false, b2, (short)0, acc2, false, false);
    acc3 = __builtin_amdgcn_wmma_f32_16x16x32_bf16(false, av, false, b3, (short)0, acc3, false, false);
  }

  f32x8 accs[4] = { acc0, acc1, acc2, acc3 };
#pragma unroll
  for (int u = 0; u < 4; ++u) {
    const int n = n0 + u * 16 + r;
    float bn = 0.f;
    if (EPI == EPI_BIAS || EPI == EPI_GELU || EPI == EPI_SOFTPLUS) bn = bias[n];
#pragma unroll
    for (int j = 0; j < 8; ++j) {
      int m = tm * 16 + j + 8 * half;
      float v = accs[u][j] + bn;
      if (EPI == EPI_GELU)     v = 0.5f * v * (1.0f + erff(v * 0.70710678f));
      if (EPI == EPI_SOFTPLUS) v = (v > 20.f) ? v : log1pf(expf(v));
      size_t idx = (size_t)m * ldc + n;
      if (EPI == EPI_RESID) v += Cf[idx];
      if (Cf) Cf[idx] = v;
      if (Cb) Cb[idx] = f2bf(v);
    }
  }
}

// ---------------- LayerNorm (one wave per row) ----------------
__global__ __launch_bounds__(256)
void ln_kernel(const float* __restrict__ x, const float* __restrict__ g,
               const float* __restrict__ b, bf16* __restrict__ out, int Mm, int Dd)
{
  int row = blockIdx.x * 8 + (threadIdx.x >> 5);
  if (row >= Mm) return;
  int lane = threadIdx.x & 31;
  const float* xr = x + (size_t)row * Dd;
  float s = 0.f, ss = 0.f;
  for (int i = lane; i < Dd; i += 32) { float v = xr[i]; s += v; ss += v * v; }
  for (int m = 16; m >= 1; m >>= 1) { s += __shfl_xor(s, m); ss += __shfl_xor(ss, m); }
  float mu  = s / Dd;
  float var = ss / Dd - mu * mu;
  float rs  = rsqrtf(var + 1e-5f);
  bf16* orow = out + (size_t)row * Dd;
  for (int i = lane; i < Dd; i += 32)
    orow[i] = f2bf((xr[i] - mu) * rs * g[i] + b[i]);
}

// ---------------- causal depthwise conv (K=4) + SiLU ----------------
__global__ __launch_bounds__(256)
void conv_silu(const float* __restrict__ xz, const float* __restrict__ w,
               const float* __restrict__ cb, float* __restrict__ xf,
               bf16* __restrict__ xb)
{
  int i = blockIdx.x * 256 + threadIdx.x;
  if (i >= Mc * DIc) return;
  int d  = i % DIc;
  int bt = i / DIc;
  int t  = bt % Tc;
  float acc = cb[d];
  const float* wd = w + d * CK;
#pragma unroll
  for (int j = 0; j < CK; ++j) {
    int tt = t - (CK - 1) + j;
    if (tt >= 0) acc += wd[j] * xz[(size_t)(bt - t + tt) * (2 * DIc) + d];
  }
  float s = acc / (1.f + expf(-acc));        // silu
  xf[i] = s;
  xb[i] = f2bf(s);
}

// ---------------- selective scan + skip + SiLU(z) gate ----------------
// lane n in each 16-lane group owns state n of channel (b,d)
__global__ __launch_bounds__(256)
void scan_gate(const float* __restrict__ dt, const float* __restrict__ dbl,
               const float* __restrict__ xz, const float* __restrict__ xf,
               const float* __restrict__ A_log, const float* __restrict__ Dp,
               bf16* __restrict__ yb)
{
  int tid = blockIdx.x * 256 + threadIdx.x;
  int n   = tid & (SN - 1);
  int ch  = tid >> 4;
  if (ch >= Bc * DIc) return;
  int d = ch % DIc;
  int b = ch / DIc;
  float A  = -expf(A_log[d * SN + n]);
  float Dd = Dp[d];
  float h  = 0.f;
  for (int t = 0; t < Tc; ++t) {
    size_t bt  = (size_t)b * Tc + t;
    float dtv = dt[bt * DIc + d];
    float xv  = xf[bt * DIc + d];
    float Bv  = dbl[bt * NRc + Rc + n];
    float Cv  = dbl[bt * NRc + Rc + SN + n];
    h = expf(dtv * A) * h + dtv * Bv * xv;
    float p = h * Cv;
    p += __shfl_xor(p, 1); p += __shfl_xor(p, 2);
    p += __shfl_xor(p, 4); p += __shfl_xor(p, 8);
    if (n == 0) {
      float zv = xz[bt * (2 * DIc) + DIc + d];
      float gz = zv / (1.f + expf(-zv));
      yb[bt * DIc + d] = f2bf((p + Dd * xv) * gz);
    }
  }
}

// ---------------- host orchestration ----------------
extern "C" void kernel_launch(void* const* d_in, const int* in_sizes, int n_in,
                              void* d_out, int out_size, void* d_ws, size_t ws_size,
                              hipStream_t stream)
{
  const float* x      = (const float*)d_in[0];
  const float* W_in   = (const float*)d_in[1];
  const float* b_in   = (const float*)d_in[2];
  const float* ln_g   = (const float*)d_in[3];
  const float* ln_b   = (const float*)d_in[4];
  const float* in_w   = (const float*)d_in[5];
  const float* conv_w = (const float*)d_in[6];
  const float* conv_b = (const float*)d_in[7];
  const float* xproj  = (const float*)d_in[8];
  const float* dt_w   = (const float*)d_in[9];
  const float* dt_b   = (const float*)d_in[10];
  const float* A_log  = (const float*)d_in[11];
  const float* Dparam = (const float*)d_in[12];
  const float* out_w  = (const float*)d_in[13];
  const float* fn_g   = (const float*)d_in[14];
  const float* fn_b   = (const float*)d_in[15];
  const float* W1     = (const float*)d_in[16];
  const float* b1     = (const float*)d_in[17];
  const float* W2     = (const float*)d_in[18];
  const float* b2     = (const float*)d_in[19];
  float* outp = (float*)d_out;

  char* p = (char*)d_ws;
  auto alloc = [&](size_t bytes) -> void* {
    void* r = (void*)p;
    p += (bytes + 255) & ~(size_t)255;
    return r;
  };

  // f32 activations
  float* h     = (float*)alloc(sizeof(float) * Mc * Dc);
  float* xzb   = (float*)alloc(sizeof(float) * Mc * 2 * DIc);
  float* xs_f  = (float*)alloc(sizeof(float) * Mc * DIc);
  float* dbl_f = (float*)alloc(sizeof(float) * Mc * NRc);
  float* dt_f  = (float*)alloc(sizeof(float) * Mc * DIc);
  // bf16 activations (GEMM operands)
  bf16* x_b   = (bf16*)alloc(2ull * Mc * Fc);
  bf16* xln_b = (bf16*)alloc(2ull * Mc * Dc);
  bf16* xs_b  = (bf16*)alloc(2ull * Mc * DIc);
  bf16* dbl_b = (bf16*)alloc(2ull * Mc * NRc);
  bf16* y_b   = (bf16*)alloc(2ull * Mc * DIc);
  bf16* hn_b  = (bf16*)alloc(2ull * Mc * Dc);
  bf16* h1_b  = (bf16*)alloc(2ull * Mc * FFc);
  // bf16 weights
  bf16* Win_b  = (bf16*)alloc(2ull * Dc * Fc);
  bf16* inw_b  = (bf16*)alloc(2ull * Lc * 2 * DIc * Dc);
  bf16* xpj_b  = (bf16*)alloc(2ull * Lc * NRc * DIc);
  bf16* dtw_b  = (bf16*)alloc(2ull * Lc * DIc * Rc);
  bf16* outw_b = (bf16*)alloc(2ull * Lc * Dc * DIc);
  bf16* W1_b   = (bf16*)alloc(2ull * FFc * Dc);
  bf16* W2_b   = (bf16*)alloc(2ull * Fc * FFc);

  auto cvt = [&](const float* src, bf16* dst, int n) {
    cvt_bf16<<<(n + 255) / 256, 256, 0, stream>>>(src, dst, n);
  };
  cvt(x, x_b, Mc * Fc);
  cvt(W_in, Win_b, Dc * Fc);
  cvt(in_w, inw_b, Lc * 2 * DIc * Dc);
  cvt(xproj, xpj_b, Lc * NRc * DIc);
  cvt(dt_w, dtw_b, Lc * DIc * Rc);
  cvt(out_w, outw_b, Lc * Dc * DIc);
  cvt(W1, W1_b, FFc * Dc);
  cvt(W2, W2_b, Fc * FFc);

  auto gblk = [](int Mm, int Nn) { return ((Mm >> 4) * (Nn >> 6) + 7) / 8; };

  // h = x @ W_in^T + b_in
  wmma_gemm<EPI_BIAS><<<gblk(Mc, Dc), 256, 0, stream>>>(
      x_b, Fc, Win_b, Fc, b_in, h, (bf16*)nullptr, Dc, Mc, Dc, Fc);

  for (int l = 0; l < Lc; ++l) {
    ln_kernel<<<(Mc + 7) / 8, 256, 0, stream>>>(h, ln_g + l * Dc, ln_b + l * Dc,
                                                xln_b, Mc, Dc);
    // xz = ln(h) @ in_w^T
    wmma_gemm<EPI_NONE><<<gblk(Mc, 2 * DIc), 256, 0, stream>>>(
        xln_b, Dc, inw_b + (size_t)l * 2 * DIc * Dc, Dc, nullptr,
        xzb, (bf16*)nullptr, 2 * DIc, Mc, 2 * DIc, Dc);
    // xs = silu(causal_conv(xz[:, :DI]))
    conv_silu<<<(Mc * DIc + 255) / 256, 256, 0, stream>>>(
        xzb, conv_w + l * DIc * CK, conv_b + l * DIc, xs_f, xs_b);
    // dbl = xs @ xproj^T  -> [dt_r | B | C]
    wmma_gemm<EPI_NONE><<<gblk(Mc, NRc), 256, 0, stream>>>(
        xs_b, DIc, xpj_b + (size_t)l * NRc * DIc, DIc, nullptr,
        dbl_f, dbl_b, NRc, Mc, NRc, DIc);
    // dt = softplus(dbl[:, :R] @ dt_w^T + dt_b)
    wmma_gemm<EPI_SOFTPLUS><<<gblk(Mc, DIc), 256, 0, stream>>>(
        dbl_b, NRc, dtw_b + (size_t)l * DIc * Rc, Rc, dt_b + l * DIc,
        dt_f, (bf16*)nullptr, DIc, Mc, DIc, Rc);
    // selective scan + D*xs + silu(z) gate  -> y (bf16)
    scan_gate<<<(Bc * DIc * SN) / 256, 256, 0, stream>>>(
        dt_f, dbl_f, xzb, xs_f, A_log + (size_t)l * DIc * SN,
        Dparam + l * DIc, y_b);
    // h += y @ out_w^T
    wmma_gemm<EPI_RESID><<<gblk(Mc, Dc), 256, 0, stream>>>(
        y_b, DIc, outw_b + (size_t)l * Dc * DIc, DIc, nullptr,
        h, (bf16*)nullptr, Dc, Mc, Dc, DIc);
  }

  // final LN + FFN
  ln_kernel<<<(Mc + 7) / 8, 256, 0, stream>>>(h, fn_g, fn_b, hn_b, Mc, Dc);
  wmma_gemm<EPI_GELU><<<gblk(Mc, FFc), 256, 0, stream>>>(
      hn_b, Dc, W1_b, Dc, b1, (float*)nullptr, h1_b, FFc, Mc, FFc, Dc);
  wmma_gemm<EPI_BIAS><<<gblk(Mc, Fc), 256, 0, stream>>>(
      h1_b, FFc, W2_b, FFc, b2, outp, (bf16*)nullptr, Fc, Mc, Fc, FFc);

  (void)in_sizes; (void)n_in; (void)out_size; (void)ws_size;
}
